// Feature_loss_57509612093444
// MI455X (gfx1250) — matrix operand order repaired and checked by reading, not compiled, not run on hardware
//
#include <hip/hip_runtime.h>
#include <hip/hip_bf16.h>
#include <math.h>

#define B_N 8192
#define D_N 128
#define C_N 1000
#define EPS_F 1e-6f
#define LAMBD_F 0.1f

typedef _Float16 v16h __attribute__((ext_vector_type(16)));
typedef _Float16 v8h  __attribute__((ext_vector_type(8)));
typedef float    v8f  __attribute__((ext_vector_type(8)));

// ---------------------------------------------------------------- init -----
__global__ void fl_init_kernel(int* focc) {
    int i = blockIdx.x * blockDim.x + threadIdx.x;
    if (i < C_N) focc[i] = 0x7fffffff;
}

// --------------------------------------------------- prep: x -> f16 + sums --
// One block (128 threads) per row. sq/s computed from the ROUNDED values so
// the diagonal of d2 stays consistent with the f16 Gram.
__global__ void fl_prep_kernel(const float* __restrict__ x,
                               _Float16* __restrict__ xh,
                               float* __restrict__ sq,
                               float* __restrict__ s) {
    __shared__ float r1[128];
    __shared__ float r2[128];
    const int row = blockIdx.x;
    const int t   = threadIdx.x;
    float v = x[(size_t)row * D_N + t];
    _Float16 h = (_Float16)v;
    xh[(size_t)row * D_N + t] = h;
    float f = (float)h;
    r1[t] = f;
    r2[t] = f * f;
    __syncthreads();
    for (int off = 64; off > 0; off >>= 1) {
        if (t < off) { r1[t] += r1[t + off]; r2[t] += r2[t + off]; }
        __syncthreads();
    }
    if (t == 0) { s[row] = r1[0]; sq[row] = r2[0]; }
}

// ------------------------------------------------- first positive indices ---
__global__ void fl_focc_kernel(const int* __restrict__ y, int* __restrict__ focc) {
    int i = blockIdx.x * blockDim.x + threadIdx.x;
    if (i < B_N) atomicMin(&focc[y[i]], i);
}

__global__ void fl_fp_kernel(const int* __restrict__ y,
                             const int* __restrict__ focc,
                             int* __restrict__ fp) {
    int i = blockIdx.x * blockDim.x + threadIdx.x;
    if (i < B_N) fp[i] = focc[y[i]];
}

// --------------------------------------------------------- cross entropy ----
// One block (256 threads) per row of data[8192,1000]; writes -logp(label).
__global__ void fl_ce_kernel(const float* __restrict__ data,
                             const int* __restrict__ y,
                             float* __restrict__ celp) {
    __shared__ float red[256];
    const int row = blockIdx.x;
    const int t   = threadIdx.x;
    const float* dr = data + (size_t)row * C_N;

    float m = -3.402823466e38f;
    for (int j = t; j < C_N; j += 256) m = fmaxf(m, dr[j]);
    red[t] = m;
    __syncthreads();
    for (int off = 128; off > 0; off >>= 1) {
        if (t < off) red[t] = fmaxf(red[t], red[t + off]);
        __syncthreads();
    }
    m = red[0];
    __syncthreads();

    float se = 0.0f;
    for (int j = t; j < C_N; j += 256) se += expf(dr[j] - m);
    red[t] = se;
    __syncthreads();
    for (int off = 128; off > 0; off >>= 1) {
        if (t < off) red[t] += red[t + off];
        __syncthreads();
    }
    if (t == 0) {
        float lp = dr[y[row]] - m - logf(red[0]);
        celp[row] = -lp;
    }
}

// ----------------------------------------- d_pos: dist(i, first_pos[i]) -----
// One wave (32 lanes) per row; 8 rows per 256-thread block. Uses the same
// f16-rounded values with f32 accumulation for consistency with the WMMA path.
__global__ void fl_dpos_kernel(const _Float16* __restrict__ xh,
                               const float* __restrict__ sq,
                               const float* __restrict__ s,
                               const int* __restrict__ fp,
                               float* __restrict__ dpos) {
    const int lane = threadIdx.x & 31;
    const int i    = (blockIdx.x * blockDim.x + threadIdx.x) >> 5;
    if (i >= B_N) return;
    const int j = fp[i];
    const _Float16* pi = xh + (size_t)i * D_N + lane * 4;
    const _Float16* pj = xh + (size_t)j * D_N + lane * 4;
    float dot = 0.0f;
#pragma unroll
    for (int k = 0; k < 4; ++k) dot += (float)pi[k] * (float)pj[k];
    dot += __shfl_xor(dot, 1);
    dot += __shfl_xor(dot, 2);
    dot += __shfl_xor(dot, 4);
    dot += __shfl_xor(dot, 8);
    dot += __shfl_xor(dot, 16);
    if (lane == 0) {
        float d2 = sq[i] + sq[j] - 2.0f * dot
                 + 2.0f * EPS_F * (s[i] - s[j]) + (float)D_N * EPS_F * EPS_F;
        dpos[i] = __builtin_amdgcn_sqrtf(fmaxf(d2, 0.0f));
    }
}

// ----------------------------------------------- WMMA pairwise distances ----
// Each wave computes a 16(M) x 128(N) strip of dist. 8 waves / block.
// Tiles: 16x16 f32 accum, K=128 via 4x v_wmma_f32_16x16x32_f16.
// Per-row strip sums -> part[nStrip*B + row] (unique writer, deterministic).
__global__ __launch_bounds__(256) void fl_dist_kernel(
        const _Float16* __restrict__ xh,
        const float* __restrict__ sq,
        const float* __restrict__ s,
        float* __restrict__ part) {   // [64][B_N]
    const int lane   = threadIdx.x & 31;
    const int wave   = threadIdx.x >> 5;
    const int gw     = blockIdx.x * 8 + wave;   // 0..32767
    const int mTile  = gw >> 6;                 // 0..511
    const int nStrip = gw & 63;                 // 0..63
    const int mBase  = mTile * 16;
    const int nBase0 = nStrip * 128;

    const int lm = lane & 15;                   // 0..15
    const int kh = lane >> 4;                   // 0 or 1

    // ---- A fragments for this wave's 16 rows, all 4 K-blocks (ISA layout:
    // lane L holds M=L&15; halves 0..7 -> K=kh*8+i, halves 8..15 -> K=16+kh*8+i)
    v16h a[4];
    {
        const _Float16* rp = xh + (size_t)(mBase + lm) * D_N;
#pragma unroll
        for (int kb = 0; kb < 4; ++kb) {
            const _Float16* p = rp + kb * 32;
            v8h lo = *(const v8h*)(p + kh * 8);
            v8h hi = *(const v8h*)(p + 16 + kh * 8);
#pragma unroll
            for (int i = 0; i < 8; ++i) { a[kb][i] = lo[i]; a[kb][8 + i] = hi[i]; }
        }
    }

    // Per-row constants: this lane covers rows rowBase..rowBase+7 (C layout:
    // VGPR v, lane L -> M = v + 8*(L>>4), N = L&15).
    const int rowBase = mBase + 8 * kh;
    float sqr[8], sr[8];
#pragma unroll
    for (int v = 0; v < 8; ++v) {
        sqr[v] = sq[rowBase + v];
        sr[v]  = s[rowBase + v];
    }

    float acc[8];
#pragma unroll
    for (int v = 0; v < 8; ++v) acc[v] = 0.0f;

    const float epsconst = (float)D_N * EPS_F * EPS_F;

    for (int tn = 0; tn < 8; ++tn) {
        const int nBase = nBase0 + tn * 16;
        v8f c = {};
#pragma unroll
        for (int kb = 0; kb < 4; ++kb) {
            // B (32x16, K-major per lane): lane L -> N=L&15, halves 0..7 -> K=kb*32+kh*8+i,
            // halves 8..15 -> K=kb*32+16+kh*8+i of row (nBase+lm) of X (same as A).
            const _Float16* p = xh + (size_t)(nBase + lm) * D_N + kb * 32;
            v8h blo = *(const v8h*)(p + kh * 8);
            v8h bhi = *(const v8h*)(p + 16 + kh * 8);
            v16h b;
#pragma unroll
            for (int i = 0; i < 8; ++i) { b[i] = blo[i]; b[8 + i] = bhi[i]; }
            c = __builtin_amdgcn_wmma_f32_16x16x32_f16(
                    false, a[kb], false, b, (short)0, c, false, false);
        }
        const int   col = nBase + lm;
        const float sqc = sq[col];
        const float sc  = s[col];
#pragma unroll
        for (int v = 0; v < 8; ++v) {
            float d2 = sqr[v] + sqc - 2.0f * c[v]
                     + 2.0f * EPS_F * (sr[v] - sc) + epsconst;
            // raw v_sqrt_f32 (~1 ulp) -- skip libm's Newton fixup sequence
            acc[v] += __builtin_amdgcn_sqrtf(fmaxf(d2, 0.0f));
        }
    }

    // Reduce each row-partial across the 16 lanes of the half-group.
#pragma unroll
    for (int v = 0; v < 8; ++v) {
        float t = acc[v];
        t += __shfl_xor(t, 1);
        t += __shfl_xor(t, 2);
        t += __shfl_xor(t, 4);
        t += __shfl_xor(t, 8);
        if (lm == 0) part[(size_t)nStrip * B_N + rowBase + v] = t;
    }
}

// ------------------------------------------------------------- finalize -----
__global__ void fl_final_kernel(const float* __restrict__ part,   // [64][B_N]
                                const float* __restrict__ dpos,
                                const float* __restrict__ celp,
                                float* __restrict__ out) {
    __shared__ float red[256];
    __shared__ float redce[256];
    const int t = threadIdx.x;
    float lf = 0.0f;
    float ce = 0.0f;
    for (int i = t; i < B_N; i += 256) {
        float dt = 0.0f;
        for (int st = 0; st < 64; ++st) dt += part[(size_t)st * B_N + i];
        float dp = dpos[i];
        lf += dp / (dt - dp);
        ce += celp[i];
    }
    red[t] = lf;
    redce[t] = ce;
    __syncthreads();
    for (int off = 128; off > 0; off >>= 1) {
        if (t < off) { red[t] += red[t + off]; redce[t] += redce[t + off]; }
        __syncthreads();
    }
    if (t == 0) {
        float loss_feature = red[0];
        float loss_acc = redce[0] / (float)B_N;
        out[0] = loss_acc + LAMBD_F * loss_feature;
        out[1] = loss_acc;
        out[2] = loss_feature;
    }
}

// ------------------------------------------------------------------ launch --
extern "C" void kernel_launch(void* const* d_in, const int* in_sizes, int n_in,
                              void* d_out, int out_size, void* d_ws, size_t ws_size,
                              hipStream_t stream) {
    const float* data = (const float*)d_in[0];   // [B, C]
    const float* x    = (const float*)d_in[1];   // [B, D]
    const int*   y    = (const int*)d_in[2];     // [B]
    float* out = (float*)d_out;                  // 3 floats

    // Workspace carve-up (~4.4 MB total).
    char* ws = (char*)d_ws;
    _Float16* xh = (_Float16*)ws;                       // B*D halfs   = 2 MB
    float* sq    = (float*)(ws + (size_t)B_N * D_N * 2);
    float* s     = sq + B_N;
    float* dpos  = s + B_N;
    int*   focc  = (int*)(dpos + B_N);
    int*   fp    = focc + 1024;                         // C padded to 1024
    float* celp  = (float*)(fp + B_N);
    float* part  = celp + B_N;                          // 64 * B floats = 2 MB

    fl_init_kernel<<<4, 256, 0, stream>>>(focc);
    fl_prep_kernel<<<B_N, 128, 0, stream>>>(x, xh, sq, s);
    fl_focc_kernel<<<(B_N + 255) / 256, 256, 0, stream>>>(y, focc);
    fl_fp_kernel<<<(B_N + 255) / 256, 256, 0, stream>>>(y, focc, fp);
    fl_ce_kernel<<<B_N, 256, 0, stream>>>(data, y, celp);
    fl_dpos_kernel<<<(B_N * 32) / 256, 256, 0, stream>>>(xh, sq, s, fp, dpos);
    fl_dist_kernel<<<4096, 256, 0, stream>>>(xh, sq, s, part);
    fl_final_kernel<<<1, 256, 0, stream>>>(part, dpos, celp, out);
}
